// VectorQuantizer2_74423193305764
// MI455X (gfx1250) — matrix operand order repaired and checked by reading, not compiled, not run on hardware
//
#include <hip/hip_runtime.h>

// ---------------------------------------------------------------------------
// VectorQuantizer for MI455X (gfx1250, wave32, WMMA).
//
// Shapes: z [8,256,8,16,16] f32 -> N=16384 rows of C=256; emb [8192,256] f32.
// dist GEMM = 16384 x 8192 x 256 -> 68.7 GFLOP, compute bound (I/O ~40MB,
// ~1.7us @ 23.3 TB/s HBM).  Split-bf16 (hi+lo) with v_wmma_f32_16x16x32_bf16,
// 3 passes (hh+hl+lh) for ~fp32 accuracy at effective K=10.7/instr vs 4 for
// the native f32 WMMA.  All split operands (24 MB) fit in the 192 MB L2 ->
// no LDS tiling; each wave keeps its 16x256 A panel in registers and streams
// B from L2.  Three independent accumulator chains (hh/hl/lh) + explicitly
// double-buffered B loads keep the XDL pipe busy at low wave occupancy.
// ---------------------------------------------------------------------------

typedef __attribute__((ext_vector_type(16))) __bf16 v16bf;
typedef __attribute__((ext_vector_type(8)))  float  v8f;

#define N_VEC   16384      // number of z vectors (8*8*16*16)
#define N_E     8192       // codebook entries
#define C_DIM   256        // channels
#define SPATIAL 2048       // 8*16*16
#define OUT_ZQ  4194304    // 8*256*8*16*16
#define OUT_BLOCKS (OUT_ZQ / 256)   // 16384 partial sums

static __device__ __forceinline__ unsigned short f32_to_bf16(float f) {
  unsigned int u = __float_as_uint(f);
  unsigned int r = u + 0x7fffu + ((u >> 16) & 1u);   // round-to-nearest-even
  return (unsigned short)(r >> 16);
}
static __device__ __forceinline__ float bf16_to_f32(unsigned short h) {
  return __uint_as_float(((unsigned int)h) << 16);
}

// --- split z (with [b,c,s] -> [n=b*2048+s, c] transpose) into bf16 hi/lo ---
__global__ void __launch_bounds__(256)
vq_split_z(const float* __restrict__ z,
           unsigned short* __restrict__ zhi,
           unsigned short* __restrict__ zlo) {
  int t = blockIdx.x * blockDim.x + threadIdx.x;     // t = n*256 + c
  if (t >= N_VEC * C_DIM) return;
  int n = t >> 8, c = t & 255;
  int b = n >> 11, s = n & 2047;
  float v = z[((size_t)(b * C_DIM + c)) * SPATIAL + s];
  unsigned short hi = f32_to_bf16(v);
  zhi[t] = hi;
  zlo[t] = f32_to_bf16(v - bf16_to_f32(hi));
}

// --- split codebook into bf16 hi/lo (already [n_e, c] row major) -----------
__global__ void __launch_bounds__(256)
vq_split_emb(const float* __restrict__ emb,
             unsigned short* __restrict__ ehi,
             unsigned short* __restrict__ elo) {
  int t = blockIdx.x * blockDim.x + threadIdx.x;
  if (t >= N_E * C_DIM) return;
  float v = emb[t];
  unsigned short hi = f32_to_bf16(v);
  ehi[t] = hi;
  elo[t] = f32_to_bf16(v - bf16_to_f32(hi));
}

// --- ||e_k||^2 in f32 -------------------------------------------------------
__global__ void __launch_bounds__(256)
vq_enorm(const float* __restrict__ emb, float* __restrict__ enorm) {
  int k = blockIdx.x * blockDim.x + threadIdx.x;
  if (k >= N_E) return;
  const float* row = emb + (size_t)k * C_DIM;
  float s = 0.f;
#pragma unroll 8
  for (int c = 0; c < C_DIM; ++c) { float e = row[c]; s += e * e; }
  enorm[k] = s;
}

// --- main WMMA argmin kernel ------------------------------------------------
// One wave per 16-row M tile.  A panel (16x256 hi+lo) register resident.
// Loop over 512 codebook N-tiles; per 32-wide K step: 3 v_wmma_f32_16x16x32_bf16
// into 3 independent accumulator chains (hh, hl, lh); B tiles double-buffered
// so the next K-step's loads are in flight while current WMMAs execute.
// score = ||e||^2 - 2*(hh+hl+lh); running (min, argmin) per lane; butterfly
// min across the 16-lane N group.
union BFV { uint4 q[2]; v16bf v; };

__global__ void __launch_bounds__(256)
vq_argmin(const unsigned short* __restrict__ zhi,
          const unsigned short* __restrict__ zlo,
          const unsigned short* __restrict__ ehi,
          const unsigned short* __restrict__ elo,
          const float* __restrict__ enorm,
          int* __restrict__ idx) {
  const int lane = threadIdx.x & 31;
  const int wave = (blockIdx.x * blockDim.x + threadIdx.x) >> 5;
  const int l    = lane & 15;      // N / M-row selector within half
  const int grp  = lane >> 4;      // which 16-lane half
  const int m0   = wave * 16;

  // A operand in this code's contiguous k order (B matches): lanes 0-15 hold K [0..15];
  // lanes 16-31 hold [16..31]; M = lane&15.  Two b128 loads per tile.
  const size_t abase = (size_t)(m0 + l) * C_DIM;
  BFV ahi[8], alo[8];
#pragma unroll
  for (int kk = 0; kk < 8; ++kk) {
    int r1 = kk * 32 + (grp ? 16 : 0);
    int r2 = r1 + 8;
    ahi[kk].q[0] = *(const uint4*)(zhi + abase + r1);
    ahi[kk].q[1] = *(const uint4*)(zhi + abase + r2);
    alo[kk].q[0] = *(const uint4*)(zlo + abase + r1);
    alo[kk].q[1] = *(const uint4*)(zlo + abase + r2);
  }

  float best[8];
  int   bidx[8];
#pragma unroll
  for (int r = 0; r < 8; ++r) { best[r] = 3.4e38f; bidx[r] = 0; }

  // B operand in the same contiguous k order as A: N = lane&15, lanes 0-15 K=0..15, 16-31 K=16..31
  // -> one contiguous 32B run per lane.
  const int kb = grp ? 16 : 0;
  for (int n0 = 0; n0 < N_E; n0 += 16) {
    const float en = enorm[n0 + l];
    const unsigned short* bh = ehi + (size_t)(n0 + l) * C_DIM + kb;
    const unsigned short* bl = elo + (size_t)(n0 + l) * C_DIM + kb;
    v8f acc_hh = {0.f, 0.f, 0.f, 0.f, 0.f, 0.f, 0.f, 0.f};
    v8f acc_hl = {0.f, 0.f, 0.f, 0.f, 0.f, 0.f, 0.f, 0.f};
    v8f acc_lh = {0.f, 0.f, 0.f, 0.f, 0.f, 0.f, 0.f, 0.f};

    // double-buffered B tiles: buffer for step kk+1 loads while step kk MACs
    BFV bhi[2], blo[2];
    bhi[0].q[0] = *(const uint4*)(bh);
    bhi[0].q[1] = *(const uint4*)(bh + 8);
    blo[0].q[0] = *(const uint4*)(bl);
    blo[0].q[1] = *(const uint4*)(bl + 8);
#pragma unroll
    for (int kk = 0; kk < 8; ++kk) {
      const int cur = kk & 1, nxt = cur ^ 1;
      if (kk < 7) {
        const unsigned short* bhn = bh + (kk + 1) * 32;
        const unsigned short* bln = bl + (kk + 1) * 32;
        bhi[nxt].q[0] = *(const uint4*)(bhn);
        bhi[nxt].q[1] = *(const uint4*)(bhn + 8);
        blo[nxt].q[0] = *(const uint4*)(bln);
        blo[nxt].q[1] = *(const uint4*)(bln + 8);
      }
      acc_hh = __builtin_amdgcn_wmma_f32_16x16x32_bf16(
          false, ahi[kk].v, false, bhi[cur].v, (short)0, acc_hh, false, false);
      acc_hl = __builtin_amdgcn_wmma_f32_16x16x32_bf16(
          false, ahi[kk].v, false, blo[cur].v, (short)0, acc_hl, false, false);
      acc_lh = __builtin_amdgcn_wmma_f32_16x16x32_bf16(
          false, alo[kk].v, false, bhi[cur].v, (short)0, acc_lh, false, false);
    }
    // C/D layout: lanes 0-15 -> M=r, lanes 16-31 -> M=8+r; N = lane&15.
#pragma unroll
    for (int r = 0; r < 8; ++r) {
      float score = en - 2.0f * (acc_hh[r] + acc_hl[r] + acc_lh[r]);
      if (score < best[r]) { best[r] = score; bidx[r] = n0 + l; }
    }
  }

  // Min-reduce across the 16 lanes of the N group (masks stay within halves).
#pragma unroll
  for (int m = 1; m < 16; m <<= 1) {
#pragma unroll
    for (int r = 0; r < 8; ++r) {
      float ov = __shfl_xor(best[r], m, 32);
      int   oi = __shfl_xor(bidx[r], m, 32);
      if (ov < best[r] || (ov == best[r] && oi < bidx[r])) {
        best[r] = ov; bidx[r] = oi;
      }
    }
  }
  if (l == 0) {
#pragma unroll
    for (int r = 0; r < 8; ++r) idx[m0 + grp * 8 + r] = bidx[r];
  }
}

// --- gather z_q (output layout == input z layout), per-block MSE partials ---
__global__ void __launch_bounds__(256)
vq_output(const float* __restrict__ z, const float* __restrict__ emb,
          const int* __restrict__ idx, float* __restrict__ out,
          float* __restrict__ partials) {
  __shared__ float sdata[256];
  int t = blockIdx.x * blockDim.x + threadIdx.x;   // linear over [b,c,s]
  float d2 = 0.f;
  if (t < OUT_ZQ) {
    int b = t >> 19;            // / (256*2048)
    int c = (t >> 11) & 255;
    int s = t & 2047;
    int n = (b << 11) + s;
    float q  = emb[(size_t)idx[n] * C_DIM + c];
    float zv = z[t];
    float diff = q - zv;
    d2 = diff * diff;
    out[t] = q;                 // straight-through forward == z_q
  }
  if (t < N_VEC) out[OUT_ZQ + 1 + t] = (float)idx[t];   // idx as float
  sdata[threadIdx.x] = d2;
  __syncthreads();
  for (int off = 128; off > 0; off >>= 1) {
    if (threadIdx.x < off) sdata[threadIdx.x] += sdata[threadIdx.x + off];
    __syncthreads();
  }
  if (threadIdx.x == 0) partials[blockIdx.x] = sdata[0];
}

// --- deterministic loss finalize (fixed-order, no atomics) ------------------
__global__ void __launch_bounds__(256)
vq_finalize(const float* __restrict__ partials, float* __restrict__ out) {
  __shared__ float sdata[256];
  float s = 0.f;
  for (int i = threadIdx.x; i < OUT_BLOCKS; i += 256) s += partials[i];
  sdata[threadIdx.x] = s;
  __syncthreads();
  for (int off = 128; off > 0; off >>= 1) {
    if (threadIdx.x < off) sdata[threadIdx.x] += sdata[threadIdx.x + off];
    __syncthreads();
  }
  // loss = (1 + BETA) * mean(diff^2), BETA = 0.25
  if (threadIdx.x == 0) out[OUT_ZQ] = sdata[0] * (1.25f / (float)OUT_ZQ);
}

extern "C" void kernel_launch(void* const* d_in, const int* in_sizes, int n_in,
                              void* d_out, int out_size, void* d_ws, size_t ws_size,
                              hipStream_t stream) {
  (void)in_sizes; (void)n_in; (void)out_size; (void)ws_size;
  const float* z   = (const float*)d_in[0];
  const float* emb = (const float*)d_in[1];
  float* out = (float*)d_out;

  char* ws = (char*)d_ws;
  unsigned short* zhi = (unsigned short*)ws;  ws += (size_t)N_VEC * C_DIM * 2;  // 8 MB
  unsigned short* zlo = (unsigned short*)ws;  ws += (size_t)N_VEC * C_DIM * 2;  // 8 MB
  unsigned short* ehi = (unsigned short*)ws;  ws += (size_t)N_E   * C_DIM * 2;  // 4 MB
  unsigned short* elo = (unsigned short*)ws;  ws += (size_t)N_E   * C_DIM * 2;  // 4 MB
  float* enorm   = (float*)ws;                ws += (size_t)N_E * 4;
  int*   idxbuf  = (int*)ws;                  ws += (size_t)N_VEC * 4;
  float* partial = (float*)ws;                                                  // 64 KB

  vq_split_z  <<<(N_VEC * C_DIM + 255) / 256, 256, 0, stream>>>(z, zhi, zlo);
  vq_split_emb<<<(N_E   * C_DIM + 255) / 256, 256, 0, stream>>>(emb, ehi, elo);
  vq_enorm    <<<(N_E + 255) / 256,           256, 0, stream>>>(emb, enorm);
  // 16384/16 = 1024 waves, 8 waves (256 threads) per block -> 128 blocks
  vq_argmin   <<<128, 256, 0, stream>>>(zhi, zlo, ehi, elo, enorm, idxbuf);
  vq_output   <<<OUT_BLOCKS, 256, 0, stream>>>(z, emb, idxbuf, out, partial);
  vq_finalize <<<1, 256, 0, stream>>>(partial, out);
}
